// PointMixerIntraSetLayerPaper_9242769621751
// MI455X (gfx1250) — matrix operand order repaired and checked by reading, not compiled, not run on hardware
//
#include <hip/hip_runtime.h>

#define EPSV 1e-5f

typedef __bf16 bf16;
typedef bf16  v16bf __attribute__((ext_vector_type(16)));
typedef float v8f   __attribute__((ext_vector_type(8)));

union AFRAG { v16bf v; unsigned u[8]; };
union BPK { unsigned u; bf16 h[2]; };

static __device__ __forceinline__ v8f vzero8() {
  v8f v;
#pragma unroll
  for (int i = 0; i < 8; ++i) v[i] = 0.0f;
  return v;
}

static __device__ __forceinline__ v8f wmma_bf16(v16bf a, v16bf b, v8f c) {
  return __builtin_amdgcn_wmma_f32_16x16x32_bf16(false, a, false, b, (short)0, c,
                                                 false, false);
}

static __device__ __forceinline__ unsigned pk2(float a, float b) {
  BPK p; p.h[0] = (bf16)a; p.h[1] = (bf16)b; return p.u;
}

// packed bf16 multiply: d.lo = a.lo*b.lo, d.hi = a.hi*b.hi
static __device__ __forceinline__ unsigned pkmul_bf16(unsigned a, unsigned b) {
  unsigned d;
  asm("v_pk_mul_bf16 %0, %1, %2" : "=v"(d) : "v"(a), "v"(b));
  return d;
}

// A fragment (16x32 bf16): lane = M row, K split across lane halves.
static __device__ __forceinline__ v16bf load_a(const bf16* lds, int row, int stride,
                                               int kb, int kvalid, int hi) {
  v16bf a;
#pragma unroll
  for (int e = 0; e < 16; ++e) {
    int k = (e < 8 ? e : e + 8) + hi * 8;
    a[e] = (k < kvalid) ? lds[row * stride + kb + k] : (bf16)0.0f;
  }
  return a;
}

// B fragment (32x16 bf16) from [N][K]-transposed LDS: lane = N col, K = (e<8 ? e : e+8) + hi*8.
static __device__ __forceinline__ v16bf load_b(const bf16* ldsT, int col, int stride,
                                               int kb, int kvalid, int hi) {
  v16bf b;
#pragma unroll
  for (int e = 0; e < 16; ++e) {
    int k = (e < 8 ? e : e + 8) + hi * 8;
    b[e] = (k < kvalid) ? ldsT[col * stride + kb + k] : (bf16)0.0f;
  }
  return b;
}

#define WAVE_WAIT_LDS() asm volatile("s_wait_dscnt 0x0" ::: "memory")

struct alignas(16) SC {          // per-block shared constants
  bf16 w01T[16][96];             // reordered; col 67 = b01 (1-driven)
  bf16 blW2[16][256];            // blW2[o][j*16+i] = blW[o,i,j]
  bf16 c2aT[64][32];             // c2a_w^T, pre-scaled by bn2a scale
  bf16 c2bT[16][64];             // c2b_w^T, pre-scaled by bn2b scale; n>=8 zero
  bf16 w03T[64][64];             // w03^T
  bf16 lp2wT[64][32];            // lp2_w^T; col 3 = lp2_b + b03 (1-driven)
  float shift_a[64];
  float c2cwT[8][8];             // [c][q]
  float lp2s[3][16], lp2bs[16];  // shrink-folded lp2 weights/bias
  float blBc[16];
  float shift_b[8], c2cb[8];
  float lp1w[12], lp1b[4], scale_p[4], shift_p[4];
};

struct alignas(16) WS {          // per-wave scratch
  bf16 xknn[16][96];             // [x(0..63) | p_r(64..66) | one(67) | zero pad]
  bf16 h[16][16];
  bf16 e[16][32];                // [energy(16) | shrink(16)]
  bf16 h2[16][64];
  float prl[16][4];
  float h3[16][8];
  float lg[8][16];               // softmax weights, [c][k]
  float red[2][64];
  int idx[16];
  int pad_[12];
};

__global__ __launch_bounds__(64) void pointmixer_kernel(
    const float* __restrict__ p, const float* __restrict__ x,
    const int* __restrict__ knn, const float* __restrict__ w01,
    const float* __restrict__ b01, const float* __restrict__ blW,
    const float* __restrict__ blB, const float* __restrict__ lp1_w,
    const float* __restrict__ lp1_b, const float* __restrict__ bnp_g,
    const float* __restrict__ bnp_b, const float* __restrict__ bnp_m,
    const float* __restrict__ bnp_v, const float* __restrict__ lp2_w,
    const float* __restrict__ lp2_b, const float* __restrict__ c2a_w,
    const float* __restrict__ bn2a_g, const float* __restrict__ bn2a_b,
    const float* __restrict__ bn2a_m, const float* __restrict__ bn2a_v,
    const float* __restrict__ c2b_w, const float* __restrict__ bn2b_g,
    const float* __restrict__ bn2b_b, const float* __restrict__ bn2b_m,
    const float* __restrict__ bn2b_v, const float* __restrict__ c2c_w,
    const float* __restrict__ c2c_b, const float* __restrict__ w03,
    const float* __restrict__ b03, float* __restrict__ out_sum,
    float* __restrict__ out_xk, int* __restrict__ out_knn,
    float* __restrict__ out_pr, int n_pts) {
  __shared__ SC sc;
  __shared__ WS wsc[2];
  const int tid = threadIdx.x;

  // ---- build shared constant images (BN/bias folded) ---------------------
  for (int q = tid; q < 16 * 96; q += 64) {
    int nn = q / 96, ii = q % 96;
    bf16 v = (bf16)0.0f;
    if (ii < 64) v = (bf16)w01[(3 + ii) * 16 + nn];
    else if (ii < 67) v = (bf16)w01[(ii - 64) * 16 + nn];
    else if (ii == 67) v = (bf16)b01[nn];
    sc.w01T[nn][ii] = v;
  }
  for (int q = tid; q < 4096; q += 64) {
    int o = q >> 8, jj = (q >> 4) & 15, ii = q & 15;
    sc.blW2[o][q & 255] = (bf16)blW[o * 256 + ii * 16 + jj];
  }
  for (int q = tid; q < 2048; q += 64) {
    int nn = q / 32, kk = q % 32;
    float s = bn2a_g[nn] * rsqrtf(bn2a_v[nn] + EPSV);
    sc.c2aT[nn][kk] = (bf16)(c2a_w[kk * 64 + nn] * s);
  }
  for (int q = tid; q < 1024; q += 64) {
    int nn = q / 64, kk = q % 64;
    bf16 v = (bf16)0.0f;
    if (nn < 8) {
      float s = bn2b_g[nn] * rsqrtf(bn2b_v[nn] + EPSV);
      v = (bf16)(c2b_w[kk * 8 + nn] * s);
    }
    sc.c2bT[nn][kk] = v;
  }
  for (int q = tid; q < 4096; q += 64) {
    int nn = q / 64, kk = q % 64;
    sc.w03T[nn][kk] = (bf16)w03[kk * 64 + nn];
  }
  for (int q = tid; q < 2048; q += 64) {
    int nn = q / 32, kk = q % 32;
    bf16 v = (bf16)0.0f;
    if (kk < 3) v = (bf16)lp2_w[kk * 64 + nn];
    else if (kk == 3) v = (bf16)(lp2_b[nn] + b03[nn]);
    sc.lp2wT[nn][kk] = v;
  }
  if (tid < 64) {
    float s = bn2a_g[tid] * rsqrtf(bn2a_v[tid] + EPSV);
    sc.shift_a[tid] = bn2a_b[tid] - bn2a_m[tid] * s;
    sc.c2cwT[tid >> 3][tid & 7] = c2c_w[(tid & 7) * 8 + (tid >> 3)];
  }
  if (tid < 16) {
    sc.blBc[tid] = blB[tid];
#pragma unroll
    for (int d = 0; d < 3; ++d)
      sc.lp2s[d][tid] = lp2_w[d * 64 + tid] + lp2_w[d * 64 + 16 + tid] +
                        lp2_w[d * 64 + 32 + tid] + lp2_w[d * 64 + 48 + tid];
    sc.lp2bs[tid] = lp2_b[tid] + lp2_b[16 + tid] + lp2_b[32 + tid] +
                    lp2_b[48 + tid];
  }
  if (tid < 9) sc.lp1w[tid] = lp1_w[tid];
  if (tid < 8) {
    float s = bn2b_g[tid] * rsqrtf(bn2b_v[tid] + EPSV);
    sc.shift_b[tid] = bn2b_b[tid] - bn2b_m[tid] * s;
    sc.c2cb[tid] = c2c_b[tid];
  }
  if (tid < 3) {
    sc.lp1b[tid] = lp1_b[tid];
    float s = bnp_g[tid] * rsqrtf(bnp_v[tid] + EPSV);
    sc.scale_p[tid] = s;
    sc.shift_p[tid] = bnp_b[tid] - bnp_m[tid] * s;
  }
  __syncthreads();

  WS& W = wsc[tid >> 5];
  const int t = tid & 31;
  const int lo = t & 15;
  const int hi = t >> 4;
  const int wave = blockIdx.x * 2 + (tid >> 5);
  const int nwaves = gridDim.x * 2;

  // ---- hoist loop-invariant fragments & scalars ---------------------------
  v16bf a_c2a[4];
#pragma unroll
  for (int m = 0; m < 4; ++m)
    a_c2a[m] = load_a(&sc.c2aT[0][0], m * 16 + lo, 32, 0, 32, hi);
  v16bf a_c2b[2];
#pragma unroll
  for (int cc = 0; cc < 2; ++cc)
    a_c2b[cc] = load_a(&sc.c2bT[0][0], lo, 64, cc * 32, 64, hi);
  const float blB_lo = sc.blBc[lo];
  float shb[8];
#pragma unroll
  for (int r = 0; r < 8; ++r) shb[r] = sc.shift_b[r];
  float l1w[9], l1b[3], spv[3], shp[3];
#pragma unroll
  for (int i = 0; i < 9; ++i) l1w[i] = sc.lp1w[i];
#pragma unroll
  for (int i = 0; i < 3; ++i) {
    l1b[i] = sc.lp1b[i]; spv[i] = sc.scale_p[i]; shp[i] = sc.shift_p[i];
  }

  for (int n = wave; n < n_pts; n += nwaves) {
    v8f pembD[4];   // p_embed (+lp2_b+b03) tiles, register-resident to stage 6
    WAVE_WAIT_LDS();
    // ---- stage 0: gather neighbors, emit p_r, stage x_knn (bf16, packed) -
    const float px = p[n * 3 + 0], py = p[n * 3 + 1], pz = p[n * 3 + 2];
    const int k2 = t >> 1, half = t & 1;
    const int idx = knn[n * 16 + k2];
    if (half == 0) {
      W.idx[k2] = idx;
      float rx = p[(size_t)idx * 3 + 0] - px;
      float ry = p[(size_t)idx * 3 + 1] - py;
      float rz = p[(size_t)idx * 3 + 2] - pz;
      *(float4*)&W.prl[k2][0] = make_float4(rx, ry, rz, 0.0f);
      size_t ob = ((size_t)n * 16 + k2) * 3;
      out_pr[ob + 0] = rx; out_pr[ob + 1] = ry; out_pr[ob + 2] = rz;
      // elements 64..67 = [rx, ry, rz, 1.0] ; the 1 drives the b01 column
      *(uint2*)&W.xknn[k2][64] = make_uint2(pk2(rx, ry), pk2(rz, 1.0f));
    } else {
      uint4 z4 = make_uint4(0u, 0u, 0u, 0u);
      *(uint2*)&W.xknn[k2][68] = make_uint2(0u, 0u);
      *(uint4*)&W.xknn[k2][72] = z4;
      *(uint4*)&W.xknn[k2][80] = z4;
      *(uint4*)&W.xknn[k2][88] = z4;
    }
    {
      const float* xr = x + (size_t)idx * 64 + half * 32;
#pragma unroll
      for (int g = 0; g < 4; ++g) {
        float4 f0 = *(const float4*)(xr + g * 8);
        float4 f1 = *(const float4*)(xr + g * 8 + 4);
        *(uint4*)&W.xknn[k2][half * 32 + g * 8] =
            make_uint4(pk2(f0.x, f0.y), pk2(f0.z, f0.w),
                       pk2(f1.x, f1.y), pk2(f1.z, f1.w));
      }
    }
    WAVE_WAIT_LDS();

    // ---- stage 1: h = relu(x_knn @ w01 + b01)  (bias via K=67 column) ----
    {
      v8f acc = vzero8();
#pragma unroll
      for (int c = 0; c < 3; ++c) {
        v16bf a = load_a(&W.xknn[0][0], lo, 96, c * 32, 96, hi);
        v16bf b = load_b(&sc.w01T[0][0], lo, 96, c * 32, 96, hi);
        acc = wmma_bf16(a, b, acc);
      }
#pragma unroll
      for (int r = 0; r < 8; ++r)
        W.h[r + hi * 8][lo] = (bf16)fmaxf(acc[r], 0.0f);
    }
    WAVE_WAIT_LDS();

    // ---- stage 2: bilinear as one K=256 GEMM, A' built via v_pk_mul_bf16 -
    {
      uint4 hA = *(const uint4*)&W.h[lo][0];
      uint4 hB = *(const uint4*)&W.h[lo][8];
      uint4 hS = *(const uint4*)&W.h[lo][hi * 8];   // this lane-half's i-factors
      unsigned hsd[4] = {hS.x, hS.y, hS.z, hS.w};
      unsigned hrow[8] = {hA.x, hA.y, hA.z, hA.w, hB.x, hB.y, hB.z, hB.w};
      v8f en = vzero8();
#pragma unroll
      for (int c = 0; c < 8; ++c) {
        unsigned uc = hrow[c];
        unsigned sa = (uc & 0xffffu) | (uc << 16);        // splat h[2c]
        unsigned sbv = (uc >> 16) | (uc & 0xffff0000u);   // splat h[2c+1]
        AFRAG a;
#pragma unroll
        for (int d = 0; d < 8; ++d)
          a.u[d] = pkmul_bf16(d < 4 ? sa : sbv, hsd[d & 3]);
        v16bf b = load_b(&sc.blW2[0][0], lo, 256, c * 32, 256, hi);
        en = wmma_bf16(a.v, b, en);
      }
#pragma unroll
      for (int r = 0; r < 8; ++r)
        W.e[r + hi * 8][lo] = (bf16)(en[r] + blB_lo);
    }

    // ---- stage 3: q (VALU) ; shrink from folded weights ; pemb via WMMA --
    {
      float rxv = W.prl[lo][0], ryv = W.prl[lo][1], rzv = W.prl[lo][2];
      float qv[3];
#pragma unroll
      for (int d = 0; d < 3; ++d) {
        float v = rxv * l1w[0 * 3 + d] + ryv * l1w[1 * 3 + d] +
                  rzv * l1w[2 * 3 + d] + l1b[d];
        qv[d] = fmaxf(v * spv[d] + shp[d], 0.0f);
      }
      // shrink[lo][b] = q . lp2s[:,b] + lp2bs[b], b = hi*8 + 0..7
      {
        float4 s0a = *(const float4*)&sc.lp2s[0][hi * 8];
        float4 s0b = *(const float4*)&sc.lp2s[0][hi * 8 + 4];
        float4 s1a = *(const float4*)&sc.lp2s[1][hi * 8];
        float4 s1b = *(const float4*)&sc.lp2s[1][hi * 8 + 4];
        float4 s2a = *(const float4*)&sc.lp2s[2][hi * 8];
        float4 s2b = *(const float4*)&sc.lp2s[2][hi * 8 + 4];
        float4 ba  = *(const float4*)&sc.lp2bs[hi * 8];
        float4 bb4 = *(const float4*)&sc.lp2bs[hi * 8 + 4];
        float w0 = qv[0] * s0a.x + qv[1] * s1a.x + qv[2] * s2a.x + ba.x;
        float w1 = qv[0] * s0a.y + qv[1] * s1a.y + qv[2] * s2a.y + ba.y;
        float w2 = qv[0] * s0a.z + qv[1] * s1a.z + qv[2] * s2a.z + ba.z;
        float w3 = qv[0] * s0a.w + qv[1] * s1a.w + qv[2] * s2a.w + ba.w;
        float w4 = qv[0] * s0b.x + qv[1] * s1b.x + qv[2] * s2b.x + bb4.x;
        float w5 = qv[0] * s0b.y + qv[1] * s1b.y + qv[2] * s2b.y + bb4.y;
        float w6 = qv[0] * s0b.z + qv[1] * s1b.z + qv[2] * s2b.z + bb4.z;
        float w7 = qv[0] * s0b.w + qv[1] * s1b.w + qv[2] * s2b.w + bb4.w;
        *(uint4*)&W.e[lo][16 + hi * 8] = make_uint4(
            pk2(w0, w1), pk2(w2, w3), pk2(w4, w5), pk2(w6, w7));
      }
      // pemb tiles = [q0 q1 q2 1] @ lp2wT  (bias folded into column 3)
      AFRAG aq;
      unsigned dq0 = pk2(qv[0], qv[1]);
      unsigned dq1 = pk2(qv[2], 1.0f);
      aq.u[0] = hi ? 0u : dq0;
      aq.u[1] = hi ? 0u : dq1;
#pragma unroll
      for (int d2 = 2; d2 < 8; ++d2) aq.u[d2] = 0u;
#pragma unroll
      for (int tt = 0; tt < 4; ++tt) {
        v16bf b = load_b(&sc.lp2wT[0][0], tt * 16 + lo, 32, 0, 32, hi);
        pembD[tt] = wmma_bf16(aq.v, b, vzero8());
      }
    }
    WAVE_WAIT_LDS();

    // ---- stage 4: h2^T tiles = c2aT-frag @ e-frag (scale pre-folded) -----
    {
      v16bf be = load_b(&W.e[0][0], lo, 32, 0, 32, hi);
#pragma unroll
      for (int m = 0; m < 4; ++m) {
        v8f d = wmma_bf16(a_c2a[m], be, vzero8());
        int cb = m * 16 + hi * 8;
        float4 t0 = *(const float4*)&sc.shift_a[cb];
        float4 t1 = *(const float4*)&sc.shift_a[cb + 4];
        float v0 = fmaxf(d[0] + t0.x, 0.0f);
        float v1 = fmaxf(d[1] + t0.y, 0.0f);
        float v2 = fmaxf(d[2] + t0.z, 0.0f);
        float v3 = fmaxf(d[3] + t0.w, 0.0f);
        float v4 = fmaxf(d[4] + t1.x, 0.0f);
        float v5 = fmaxf(d[5] + t1.y, 0.0f);
        float v6 = fmaxf(d[6] + t1.z, 0.0f);
        float v7 = fmaxf(d[7] + t1.w, 0.0f);
        *(uint4*)&W.h2[lo][cb] = make_uint4(pk2(v0, v1), pk2(v2, v3),
                                            pk2(v4, v5), pk2(v6, v7));
      }
    }
    WAVE_WAIT_LDS();

    // ---- stage 5: h3^T = c2bT-frag @ h2-frag (scale pre-folded) ----------
    {
      v8f d5 = vzero8();
#pragma unroll
      for (int cc = 0; cc < 2; ++cc) {
        v16bf bh = load_b(&W.h2[0][0], lo, 64, cc * 32, 64, hi);
        d5 = wmma_bf16(a_c2b[cc], bh, d5);
      }
      if (hi == 0) {
        float v0 = fmaxf(d5[0] + shb[0], 0.0f);
        float v1 = fmaxf(d5[1] + shb[1], 0.0f);
        float v2 = fmaxf(d5[2] + shb[2], 0.0f);
        float v3 = fmaxf(d5[3] + shb[3], 0.0f);
        float v4 = fmaxf(d5[4] + shb[4], 0.0f);
        float v5 = fmaxf(d5[5] + shb[5], 0.0f);
        float v6 = fmaxf(d5[6] + shb[6], 0.0f);
        float v7 = fmaxf(d5[7] + shb[7], 0.0f);
        *(float4*)&W.h3[lo][0] = make_float4(v0, v1, v2, v3);
        *(float4*)&W.h3[lo][4] = make_float4(v4, v5, v6, v7);
      }
    }
    WAVE_WAIT_LDS();

    // ---- stage 5b: logits = h3 @ c2c_w + b ; softmax over k --------------
    {
      float4 ha = *(const float4*)&W.h3[lo][0];
      float4 hb = *(const float4*)&W.h3[lo][4];
      float4 cb4 = *(const float4*)&sc.c2cb[hi * 4];
      float cbv[4] = {cb4.x, cb4.y, cb4.z, cb4.w};
#pragma unroll
      for (int cc = 0; cc < 4; ++cc) {
        int c = hi * 4 + cc;
        float4 w0 = *(const float4*)&sc.c2cwT[c][0];
        float4 w1 = *(const float4*)&sc.c2cwT[c][4];
        float l = cbv[cc] + ha.x * w0.x + ha.y * w0.y + ha.z * w0.z +
                  ha.w * w0.w + hb.x * w1.x + hb.y * w1.y + hb.z * w1.z +
                  hb.w * w1.w;
        W.lg[c][lo] = l;
      }
    }
    WAVE_WAIT_LDS();
    if (t < 8) {
      float4 v0 = *(const float4*)&W.lg[t][0];
      float4 v1 = *(const float4*)&W.lg[t][4];
      float4 v2 = *(const float4*)&W.lg[t][8];
      float4 v3 = *(const float4*)&W.lg[t][12];
      float va[16] = {v0.x, v0.y, v0.z, v0.w, v1.x, v1.y, v1.z, v1.w,
                      v2.x, v2.y, v2.z, v2.w, v3.x, v3.y, v3.z, v3.w};
      float m = va[0];
#pragma unroll
      for (int kk = 1; kk < 16; ++kk) m = fmaxf(m, va[kk]);
      float s = 0.0f;
#pragma unroll
      for (int kk = 0; kk < 16; ++kk) { va[kk] = __expf(va[kk] - m); s += va[kk]; }
      float inv = 1.0f / s;
#pragma unroll
      for (int kk = 0; kk < 16; ++kk) va[kk] *= inv;
      *(float4*)&W.lg[t][0]  = make_float4(va[0], va[1], va[2], va[3]);
      *(float4*)&W.lg[t][4]  = make_float4(va[4], va[5], va[6], va[7]);
      *(float4*)&W.lg[t][8]  = make_float4(va[8], va[9], va[10], va[11]);
      *(float4*)&W.lg[t][12] = make_float4(va[12], va[13], va[14], va[15]);
    }
    WAVE_WAIT_LDS();

    // ---- stage 6: x_v tiles + register pemb ; blend ; reduce -------------
    {
      v16bf a0 = load_a(&W.xknn[0][0], lo, 96, 0, 96, hi);
      v16bf a1 = load_a(&W.xknn[0][0], lo, 96, 32, 96, hi);
      float4 lgl = *(const float4*)&W.lg[lo & 7][hi * 8];
      float4 lgh = *(const float4*)&W.lg[lo & 7][hi * 8 + 4];
      float wv[8] = {lgl.x, lgl.y, lgl.z, lgl.w, lgh.x, lgh.y, lgh.z, lgh.w};
#pragma unroll
      for (int tt = 0; tt < 4; ++tt) {
        int c = tt * 16 + lo;
        v8f d = wmma_bf16(a0, load_b(&sc.w03T[0][0], c, 64, 0, 64, hi), vzero8());
        d = wmma_bf16(a1, load_b(&sc.w03T[0][0], c, 64, 32, 64, hi), d);
        float wsum = 0.0f;
#pragma unroll
        for (int r = 0; r < 8; ++r) {
          int kk = r + hi * 8;
          float val = (d[r] + pembD[tt][r]) * wv[r];
          out_xk[((size_t)n * 16 + kk) * 64 + c] = val;
          wsum += val;
        }
        W.red[hi][c] = wsum;
      }
    }
    WAVE_WAIT_LDS();
#pragma unroll
    for (int j2 = 0; j2 < 2; ++j2) {
      int c = t + j2 * 32;
      out_sum[(size_t)n * 64 + c] = W.red[0][c] + W.red[1][c];
    }
    if (t < 16) out_knn[(size_t)n * 16 + t] = W.idx[t];
  }
}

extern "C" void kernel_launch(void* const* d_in, const int* in_sizes, int n_in,
                              void* d_out, int out_size, void* d_ws, size_t ws_size,
                              hipStream_t stream) {
  (void)n_in; (void)out_size; (void)d_ws; (void)ws_size;
  const float* p     = (const float*)d_in[0];
  const float* x     = (const float*)d_in[1];
  const int*   knn   = (const int*)d_in[2];
  const float* w01   = (const float*)d_in[3];
  const float* b01   = (const float*)d_in[4];
  const float* blW   = (const float*)d_in[5];
  const float* blB   = (const float*)d_in[6];
  const float* lp1w  = (const float*)d_in[7];
  const float* lp1b  = (const float*)d_in[8];
  const float* bnpg  = (const float*)d_in[9];
  const float* bnpb  = (const float*)d_in[10];
  const float* bnpm  = (const float*)d_in[11];
  const float* bnpv  = (const float*)d_in[12];
  const float* lp2w  = (const float*)d_in[13];
  const float* lp2b  = (const float*)d_in[14];
  const float* c2aw  = (const float*)d_in[15];
  const float* bn2ag = (const float*)d_in[16];
  const float* bn2ab = (const float*)d_in[17];
  const float* bn2am = (const float*)d_in[18];
  const float* bn2av = (const float*)d_in[19];
  const float* c2bw  = (const float*)d_in[20];
  const float* bn2bg = (const float*)d_in[21];
  const float* bn2bb = (const float*)d_in[22];
  const float* bn2bm = (const float*)d_in[23];
  const float* bn2bv = (const float*)d_in[24];
  const float* c2cw  = (const float*)d_in[25];
  const float* c2cb  = (const float*)d_in[26];
  const float* w03   = (const float*)d_in[27];
  const float* b03   = (const float*)d_in[28];

  const int n_pts = in_sizes[0] / 3;
  float* out0 = (float*)d_out;
  float* oxk  = out0 + (size_t)n_pts * 64;
  int*   oknn = (int*)(out0 + (size_t)n_pts * 64 + (size_t)n_pts * 1024);
  float* opr  = out0 + (size_t)n_pts * 64 + (size_t)n_pts * 1024 + (size_t)n_pts * 16;

  dim3 grid(4096), block(64);
  pointmixer_kernel<<<grid, block, 0, stream>>>(
      p, x, knn, w01, b01, blW, blB, lp1w, lp1b, bnpg, bnpb, bnpm, bnpv, lp2w,
      lp2b, c2aw, bn2ag, bn2ab, bn2am, bn2av, c2bw, bn2bg, bn2bb, bn2bm, bn2bv,
      c2cw, c2cb, w03, b03, out0, oxk, oknn, opr, n_pts);
}
